// ItemsNeighborsEmbeddingsAggregation_21199958573415
// MI455X (gfx1250) — compile-verified, matrix operand order checked by reading, not run on hardware
//
#include <hip/hip_runtime.h>
#include <hip/hip_bf16.h>

typedef __attribute__((ext_vector_type(16))) __bf16       v16bf;
typedef __attribute__((ext_vector_type(8)))  float        v8f;
typedef __attribute__((ext_vector_type(8)))  unsigned int v8u;
typedef __attribute__((ext_vector_type(4)))  unsigned int v4u;
typedef __attribute__((ext_vector_type(2)))  unsigned int v2u;

// ---------------- constants ----------------
static constexpr int BATCH = 16384;
static constexpr int KNB   = 20;      // neighbors
static constexpr int DIMD  = 128;
static constexpr int DIME  = 256;     // D + T
static constexpr int DIMKD = 384;     // 2D + T
static constexpr float QSCALE = 0.08838834764831845f; // 1/sqrt(128)

// bf16 <-> f32 (round to nearest even)
__device__ __forceinline__ unsigned short f2bf(float f) {
  unsigned u = __builtin_bit_cast(unsigned, f);
  u = (u + 0x7FFFu + ((u >> 16) & 1u)) >> 16;
  return (unsigned short)u;
}
__device__ __forceinline__ float bf2f(unsigned short h) {
  unsigned u = ((unsigned)h) << 16;
  return __builtin_bit_cast(float, u);
}

// K index inside a 16-bit 16x32 A/B fragment (ISA 7.12.2): dword j, lane-half, elem e
__device__ __forceinline__ int frag_k(int j, int half, int e) {
  return ((j < 4) ? (half * 8 + j * 2) : (16 + half * 8 + (j - 4) * 2)) + e;
}

// Async memory->LDS copy, 16B per lane (ASYNCcnt-tracked; gfx1250 §15.18.3 op 98).
__device__ __forceinline__ void async_ld_b128(void* lds_dst, const void* gsrc) {
  unsigned       loff = (unsigned)(uintptr_t)lds_dst;        // low 32b of generic = LDS offset
  unsigned long long ga = (unsigned long long)(uintptr_t)gsrc;
  asm volatile("global_load_async_to_lds_b128 %0, %1, off"
               :: "v"(loff), "v"(ga) : "memory");
}
__device__ __forceinline__ void wait_async0() {
  asm volatile("s_wait_asynccnt 0" ::: "memory");
}

// ---------------- weight pre-pack kernels ----------------
// Packed layout: tile(kt,nt) of a [Kd][Nd] weight -> 512 bf16 at ((nt*ktn + kt) << 9);
// within tile: lane*16 + j*2 + e   (B-fragment order, 32B contiguous per lane)
__global__ void pack_w(const float* __restrict__ src, unsigned short* __restrict__ dst,
                       int Kd, int Nd) {
  int p = blockIdx.x * blockDim.x + threadIdx.x;
  int total = Kd * Nd;
  if (p >= total) return;
  int ktn = Kd >> 5;
  int tile = p >> 9, r = p & 511;
  int lane = r >> 4, t2 = r & 15, j = t2 >> 1, e = t2 & 1;
  int nt = tile / ktn, kt = tile - nt * ktn;
  int half = lane >> 4;
  int n = (nt << 4) + (lane & 15);
  int k = (kt << 5) + frag_k(j, half, e);
  dst[p] = f2bf(src[k * Nd + n]);
}

// WkT per head: logical [128][384], elem(d,c) = Wk[c*256 + h*128 + d]
__global__ void pack_wkT(const float* __restrict__ Wk, unsigned short* __restrict__ dst) {
  int p = blockIdx.x * blockDim.x + threadIdx.x;
  if (p >= 2 * 128 * 384) return;
  int h = p / 49152, q = p - h * 49152;
  int ktn = 4;
  int tile = q >> 9, r = q & 511;
  int lane = r >> 4, t2 = r & 15, j = t2 >> 1, e = t2 & 1;
  int nt = tile / ktn, kt = tile - nt * ktn;
  int half = lane >> 4;
  int c = (nt << 4) + (lane & 15);        // 0..383
  int d = (kt << 5) + frag_k(j, half, e); // 0..127
  dst[p] = f2bf(Wk[c * 256 + h * 128 + d]);
}

// ---------------- fragment loaders ----------------
__device__ __forceinline__ v16bf ldsA(const unsigned short* lds, int S, int k0, int lane) {
  int m = lane & 15, half = lane >> 4;
  const unsigned short* p0 = lds + m * S + k0 + 8 * half;
  v4u lo = *(const v4u*)p0;
  v4u hi = *(const v4u*)(p0 + 16);
  v8u r;
  r[0] = lo[0]; r[1] = lo[1]; r[2] = lo[2]; r[3] = lo[3];
  r[4] = hi[0]; r[5] = hi[1]; r[6] = hi[2]; r[7] = hi[3];
  return __builtin_bit_cast(v16bf, r);
}
__device__ __forceinline__ v16bf ldB(const unsigned short* wp, int ktn, int kt, int nt, int lane) {
  const unsigned short* p = wp + (((size_t)(nt * ktn + kt)) << 9) + (lane << 4);
  v8u r = *(const v8u*)p;
  return __builtin_bit_cast(v16bf, r);
}
__device__ __forceinline__ v8f wmma_bf16(v16bf a, v16bf b, v8f c) {
  return __builtin_amdgcn_wmma_f32_16x16x32_bf16(false, a, false, b, (short)0, c, false, false);
}

// ---------------- LDS pool layout (bytes) ----------------
static constexpr int OFF_FEAT  = 0;       // f32 [16][128]
static constexpr int OFF_QUERY = 8192;    // u16 [16][256]  (query; later attn_out bf16)
static constexpr int OFF_QBF   = 16384;   // u16 [16][256]  (q); later sCat u16[16][384] (spans into U)
static constexpr int OFF_U     = 24576;   // u16 [16][768]
static constexpr int OFF_WBF   = 49152;   // u16 [16][768]
static constexpr int OFF_KV    = 73728;   // 2x f32 [20][384] (double buffer); later AOF32 / H1 alias
static constexpr int KVBYTES   = 30720;
static constexpr int OFF_SMALL = 135168;
static constexpr int SMEM_BYTES = 137360;

__global__ __launch_bounds__(256) void
tgat_main(const int* __restrict__ items_idxs, const float* __restrict__ timestamps,
          const int* __restrict__ inv_idx, const int* __restrict__ vot_idx,
          const int* __restrict__ edge_idx, const float* __restrict__ edge_ts,
          const float* __restrict__ items_static, const float* __restrict__ items_mem,
          const float* __restrict__ users_mem,
          const float* __restrict__ time_w, const float* __restrict__ time_b,
          const float* __restrict__ bq, const float* __restrict__ bv,
          const float* __restrict__ bo, const float* __restrict__ fc1b,
          const float* __restrict__ fc2b,
          const unsigned short* __restrict__ wq_p, const unsigned short* __restrict__ wkt_p,
          const unsigned short* __restrict__ wv_p, const unsigned short* __restrict__ wo_p,
          const unsigned short* __restrict__ f1_p, const unsigned short* __restrict__ f2_p,
          float* __restrict__ out) {
  __shared__ __align__(16) unsigned char smem[SMEM_BYTES];
  float*          sFeat  = (float*)(smem + OFF_FEAT);
  unsigned short* sQuery = (unsigned short*)(smem + OFF_QUERY);
  unsigned short* sAobf  = (unsigned short*)(smem + OFF_QUERY);  // alias
  unsigned short* sQbf   = (unsigned short*)(smem + OFF_QBF);
  unsigned short* sCat   = (unsigned short*)(smem + OFF_QBF);    // alias [16][384]
  unsigned short* sU     = (unsigned short*)(smem + OFF_U);      // [16][768]
  unsigned short* sWbf   = (unsigned short*)(smem + OFF_WBF);    // [16][768]
  float*          sAOF   = (float*)(smem + OFF_KV);              // [16][256] alias (post-loop)
  unsigned short* sH1    = (unsigned short*)(smem + OFF_KV);     // [16][128] alias (post-loop)
  int*      sIdx  = (int*)(smem + OFF_SMALL);
  float*    sTs   = (float*)(smem + OFF_SMALL + 64);
  unsigned* sMask = (unsigned*)(smem + OFF_SMALL + 128);
  int*      sInv  = (int*)(smem + OFF_SMALL + 192);
  float*    sCosB = (float*)(smem + OFF_SMALL + 256);   // 128 f32
  float*    sTW   = (float*)(smem + OFF_SMALL + 768);   // 128 f32
  float*    sTB   = (float*)(smem + OFF_SMALL + 1280);  // 128 f32
  float*    sScor = (float*)(smem + OFF_SMALL + 1792);  // 40 f32
  float*    sAttn = (float*)(smem + OFF_SMALL + 1952);  // 40 f32

  const int tid = threadIdx.x;
  const int wv = tid >> 5, lane = tid & 31;
  const int half = lane >> 4;
  const int b0 = blockIdx.x * 16;

  // Issue the kv gathers for item i into kv double-buffer (i&1) as async LDS DMA:
  // 40 rows (20 inviter + 20 voter), 512B each = one wave * b128/lane; 5 rows per wave.
  auto issue_kv = [&](int i) {
    int b = b0 + i;
    float* dst = (float*)(smem + OFF_KV + (i & 1) * KVBYTES);
    for (int task = wv; task < 2 * KNB; task += 8) {
      int table = task / KNB, r = task - table * KNB;
      int idx = table ? vot_idx[b * KNB + r] : inv_idx[b * KNB + r];
      const float* src = users_mem + (size_t)idx * DIMD + lane * 4;
      async_ld_b128(dst + r * DIMKD + table * DIMD + lane * 4, src);
    }
  };

  // Kick off item 0's gathers immediately; they overlap phases 0/1 and the Q/U GEMMs.
  issue_kv(0);

  // ---- phase 0: item meta, mask bits, time vectors ----
  if (tid < 16) {
    int b = b0 + tid;
    sIdx[tid] = items_idxs[b];
    sTs[tid]  = timestamps[b];
    unsigned mb = 0;
    for (int kk = 0; kk < KNB; ++kk)
      if (edge_idx[b * KNB + kk] == 0) mb |= (1u << kk);
    int inval = (mb == 0xFFFFFu);
    if (inval) mb &= ~(1u << 19);
    sMask[tid] = mb;
    sInv[tid]  = inval;
  }
  if (tid < 128) {
    float tb = time_b[tid];
    sCosB[tid] = __cosf(tb);
    sTB[tid]   = tb;
    sTW[tid]   = time_w[tid];
  }
  __syncthreads();

  // ---- phase 1: items_feat + query (bf16), float4 per lane, one wave per 2 rows ----
  for (int r = wv * 2; r < wv * 2 + 2; ++r) {
    int idx = sIdx[r];
    int c = lane * 4;
    float4 a = *(const float4*)(items_static + (size_t)idx * DIMD + c);
    float4 m = *(const float4*)(items_mem + (size_t)idx * DIMD + c);
    float4 f; f.x = a.x + m.x; f.y = a.y + m.y; f.z = a.z + m.z; f.w = a.w + m.w;
    *(float4*)(sFeat + r * DIMD + c) = f;
    v2u pf; pf[0] = f2bf(f.x) | ((unsigned)f2bf(f.y) << 16);
    pf[1] = f2bf(f.z) | ((unsigned)f2bf(f.w) << 16);
    *(v2u*)(sQuery + r * DIME + c) = pf;
    float4 t = *(const float4*)(sCosB + c);
    v2u pt; pt[0] = f2bf(t.x) | ((unsigned)f2bf(t.y) << 16);
    pt[1] = f2bf(t.z) | ((unsigned)f2bf(t.w) << 16);
    *(v2u*)(sQuery + r * DIME + DIMD + c) = pt;
  }
  __syncthreads();

  // ---- G1: q = (query @ Wq + bq) * 1/sqrt(hd) -> sQbf ----
  for (int t = 0; t < 2; ++t) {
    int nt = wv * 2 + t;
    int n = (nt << 4) + (lane & 15);
    float bias = bq[n];
    v8f acc; for (int j = 0; j < 8; ++j) acc[j] = bias;
    for (int kt = 0; kt < 8; ++kt)
      acc = wmma_bf16(ldsA(sQuery, DIME, kt * 32, lane), ldB(wq_p, 8, kt, nt, lane), acc);
    for (int j = 0; j < 8; ++j) {
      int m = j + 8 * half;
      sQbf[m * DIME + n] = f2bf(acc[j] * QSCALE);
    }
  }
  __syncthreads();

  // ---- G2: u_h = q_h @ Wk_h^T -> sU [16][2*384] bf16 ----
  for (int g = wv; g < 48; g += 8) {
    int h = g / 24, nt = g - h * 24;
    int n = (nt << 4) + (lane & 15);
    v8f acc = {};
    for (int kt = 0; kt < 4; ++kt)
      acc = wmma_bf16(ldsA(sQbf, DIME, h * 128 + kt * 32, lane),
                      ldB(wkt_p + h * 49152, 4, kt, nt, lane), acc);
    for (int j = 0; j < 8; ++j) {
      int m = j + 8 * half;
      sU[m * 768 + h * DIMKD + n] = f2bf(acc[j]);
    }
  }
  __syncthreads();

  // ---- per-item loop: async kv double-buffer; scores, softmax, w ----
  for (int i = 0; i < 16; ++i) {
    int b = b0 + i;
    float* kv = (float*)(smem + OFF_KV + (i & 1) * KVBYTES);
    // time-encoding third of kv (cols 256..383) is disjoint from the async-written
    // region (cols 0..255): fill it before draining ASYNCcnt.
    for (int e = tid; e < KNB * 128; e += 256) {
      int r = e >> 7, c2 = e & 127;
      float dt = sTs[i] - edge_ts[b * KNB + r];
      kv[r * DIMKD + 256 + c2] = __cosf(dt * sTW[c2] + sTB[c2]);
    }
    wait_async0();
    __syncthreads();           // kv[i&1] fully visible to all waves
    if (i < 15) issue_kv(i + 1);  // prefetch next item into the other buffer

    // scores: 2 heads x 20 neighbors, one wave per dot over KD=384
    for (int task = wv; task < 2 * KNB; task += 8) {
      int h = task / KNB, kn = task - h * KNB;
      float part = 0.f;
      for (int c = lane; c < DIMKD; c += 32)
        part += kv[kn * DIMKD + c] * bf2f(sU[i * 768 + h * DIMKD + c]);
      for (int off = 16; off; off >>= 1) part += __shfl_xor(part, off, 32);
      if (lane == 0) {
        bool masked = (sMask[i] >> kn) & 1u;
        sScor[task] = masked ? -1e9f : part;
      }
    }
    __syncthreads();
    // softmax over 20 per head (waves 0,1)
    if (wv < 2) {
      float s = (lane < KNB) ? sScor[wv * KNB + lane] : -1e30f;
      float mx = s;
      for (int off = 16; off; off >>= 1) mx = fmaxf(mx, __shfl_xor(mx, off, 32));
      float ex = (lane < KNB) ? __expf(s - mx) : 0.f;
      float sm = ex;
      for (int off = 16; off; off >>= 1) sm += __shfl_xor(sm, off, 32);
      if (lane < KNB) sAttn[wv * KNB + lane] = ex / sm;
    }
    __syncthreads();
    // w_h[c] = sum_kn attn * kv -> bf16 rows of sWbf
    for (int e = tid; e < 2 * DIMKD; e += 256) {
      int h = e / DIMKD, c = e - h * DIMKD;
      float acc = 0.f;
      for (int kn = 0; kn < KNB; ++kn) acc += sAttn[h * KNB + kn] * kv[kn * DIMKD + c];
      sWbf[i * 768 + e] = f2bf(acc);
    }
    __syncthreads();
  }

  // ---- G3: out_h = w_h @ Wv_h + bv_h -> sAOF f32 [16][256] ----
  for (int g = wv; g < 16; g += 8) {
    int h = g >> 3, nt = g;
    int n = (nt << 4) + (lane & 15);
    float bias = bv[n];
    v8f acc; for (int j = 0; j < 8; ++j) acc[j] = bias;
    for (int kt = 0; kt < 12; ++kt)
      acc = wmma_bf16(ldsA(sWbf, 768, h * DIMKD + kt * 32, lane),
                      ldB(wv_p, 12, kt, nt, lane), acc);
    for (int j = 0; j < 8; ++j) {
      int m = j + 8 * half;
      sAOF[m * DIME + n] = acc[j];
    }
  }
  __syncthreads();

  // attn_out -> bf16 (alias of sQuery), and fill feat half of sCat
  for (int e = tid; e < 16 * DIME; e += 256) sAobf[e] = f2bf(sAOF[e]);
  for (int e = tid; e < 16 * DIMD; e += 256) {
    int m = e >> 7, c = e & 127;
    sCat[m * DIMKD + DIME + c] = f2bf(sFeat[m * DIMD + c]);
  }
  __syncthreads();

  // ---- G4: out = attn_out @ Wo + bo (zero invalid) -> sCat cols [0,256) ----
  for (int t = 0; t < 2; ++t) {
    int nt = wv * 2 + t;
    int n = (nt << 4) + (lane & 15);
    float bias = bo[n];
    v8f acc; for (int j = 0; j < 8; ++j) acc[j] = bias;
    for (int kt = 0; kt < 8; ++kt)
      acc = wmma_bf16(ldsA(sAobf, DIME, kt * 32, lane), ldB(wo_p, 8, kt, nt, lane), acc);
    for (int j = 0; j < 8; ++j) {
      int m = j + 8 * half;
      float v = sInv[m] ? 0.f : acc[j];
      sCat[m * DIMKD + n] = f2bf(v);
    }
  }
  __syncthreads();

  // ---- G5: h1 = relu(cat @ fc1_w + fc1_b) -> sH1 bf16 [16][128] ----
  {
    int nt = wv;
    int n = (nt << 4) + (lane & 15);
    float bias = fc1b[n];
    v8f acc; for (int j = 0; j < 8; ++j) acc[j] = bias;
    for (int kt = 0; kt < 12; ++kt)
      acc = wmma_bf16(ldsA(sCat, DIMKD, kt * 32, lane), ldB(f1_p, 12, kt, nt, lane), acc);
    for (int j = 0; j < 8; ++j) {
      int m = j + 8 * half;
      sH1[m * DIMD + n] = f2bf(fmaxf(acc[j], 0.f));
    }
  }
  __syncthreads();

  // ---- G6: result = h1 @ fc2_w + fc2_b -> global f32 ----
  {
    int nt = wv;
    int n = (nt << 4) + (lane & 15);
    float bias = fc2b[n];
    v8f acc; for (int j = 0; j < 8; ++j) acc[j] = bias;
    for (int kt = 0; kt < 4; ++kt)
      acc = wmma_bf16(ldsA(sH1, DIMD, kt * 32, lane), ldB(f2_p, 4, kt, nt, lane), acc);
    for (int j = 0; j < 8; ++j) {
      int m = j + 8 * half;
      out[(size_t)(b0 + m) * DIMD + n] = acc[j];
    }
  }
}

// ---------------- host launch ----------------
extern "C" void kernel_launch(void* const* d_in, const int* in_sizes, int n_in,
                              void* d_out, int out_size, void* d_ws, size_t ws_size,
                              hipStream_t stream) {
  (void)in_sizes; (void)n_in; (void)out_size; (void)ws_size;
  const int*   items_idxs = (const int*)d_in[0];
  const float* timestamps = (const float*)d_in[1];
  const int*   inv_idx    = (const int*)d_in[2];
  const int*   vot_idx    = (const int*)d_in[3];
  const int*   edge_idx   = (const int*)d_in[4];
  const float* edge_ts    = (const float*)d_in[5];
  const float* items_sf   = (const float*)d_in[6];
  const float* items_mem  = (const float*)d_in[7];
  const float* users_mem  = (const float*)d_in[8];
  const float* time_w     = (const float*)d_in[9];
  const float* time_b     = (const float*)d_in[10];
  const float* Wq  = (const float*)d_in[11];
  const float* bq  = (const float*)d_in[12];
  const float* Wk  = (const float*)d_in[13];
  const float* Wv  = (const float*)d_in[15];
  const float* bv  = (const float*)d_in[16];
  const float* Wo  = (const float*)d_in[17];
  const float* bo  = (const float*)d_in[18];
  const float* f1w = (const float*)d_in[19];
  const float* f1b = (const float*)d_in[20];
  const float* f2w = (const float*)d_in[21];
  const float* f2b = (const float*)d_in[22];

  unsigned short* ws    = (unsigned short*)d_ws;
  unsigned short* wq_p  = ws;             // 65536
  unsigned short* wkt_p = ws + 65536;     // 98304 (2 heads x [128][384])
  unsigned short* wv_p  = ws + 163840;    // 98304
  unsigned short* wo_p  = ws + 262144;    // 65536
  unsigned short* f1_p  = ws + 327680;    // 49152
  unsigned short* f2_p  = ws + 376832;    // 16384

  pack_w  <<<65536 / 256, 256, 0, stream>>>(Wq, wq_p, 256, 256);
  pack_wkT<<<98304 / 256, 256, 0, stream>>>(Wk, wkt_p);
  pack_w  <<<98304 / 256, 256, 0, stream>>>(Wv, wv_p, 384, 256);
  pack_w  <<<65536 / 256, 256, 0, stream>>>(Wo, wo_p, 256, 256);
  pack_w  <<<49152 / 256, 256, 0, stream>>>(f1w, f1_p, 384, 128);
  pack_w  <<<16384 / 256, 256, 0, stream>>>(f2w, f2_p, 128, 128);

  tgat_main<<<BATCH / 16, 256, 0, stream>>>(
      items_idxs, timestamps, inv_idx, vot_idx, edge_idx, edge_ts,
      items_sf, items_mem, users_mem, time_w, time_b,
      bq, bv, bo, f1b, f2b,
      wq_p, wkt_p, wv_p, wo_p, f1_p, f2_p,
      (float*)d_out);
}
